// Generator_29454885716476
// MI455X (gfx1250) — matrix operand order repaired and checked by reading, not compiled, not run on hardware
//
#include <hip/hip_runtime.h>
#include <hip/hip_bf16.h>
#include <stdint.h>

// ---------------------------------------------------------------------------
// Neural SDE (Euler-Maruyama, 50 steps) with bf16 WMMA GEMMs, fp32 state +
// bf16 shadow state (converted once per step in the epilogue, not in GEMMs).
// B=1024, H=1024, D=64, FH=128, L=2, N_STEPS=50, DT=0.02
// ---------------------------------------------------------------------------

typedef __attribute__((ext_vector_type(16))) __bf16       v16bf;
typedef __attribute__((ext_vector_type(8)))  float        v8f;
typedef __attribute__((ext_vector_type(4)))  unsigned int uint4v;

#define B_DIM   1024
#define H_DIM   1024
#define D_DIM   64
#define FH_DIM  128
#define T_DIM   50
#define NSTEPS  50
#define DT_C       0.02f
#define SQRT_DT_C  0.141421356237309505f

union Frag {
  v16bf  v;
  uint4v q[2];
};

// A fragment (16x32, row-major bf16 source, element stride ld).
// ISA layout: lane<16 holds row=lane, K in {half*8..half*8+7, 16+half*8..+7}
__device__ __forceinline__ void load_a(Frag& f, const __bf16* base, int ld,
                                       int lane) {
  const int half = lane >> 4, r = lane & 15;
  const __bf16* p = base + (size_t)r * ld + half * 8;
  f.q[0] = *(const uint4v*)(p);
  f.q[1] = *(const uint4v*)(p + 16);
}

// B fragment (32x16) from W^T stored [N,K] bf16: lane<16 holds col n=lane,
// K in {half*8..half*8+7, 16+half*8..+7}.
__device__ __forceinline__ void load_b(Frag& f, const __bf16* baseT, int ld,
                                       int lane) {
  const int half = lane >> 4, n = lane & 15;
  const __bf16* p = baseT + (size_t)n * ld + half * 8;
  f.q[0] = *(const uint4v*)(p);
  f.q[1] = *(const uint4v*)(p + 16);
}

__device__ __forceinline__ v8f wmma_bf16(const Frag& a, const Frag& b, v8f c) {
  return __builtin_amdgcn_wmma_f32_16x16x32_bf16(false, a.v, false, b.v,
                                                 (short)0, c, false, false);
}

// ---------------------------------------------------------------------------
// Generic GEMM: D = epilogue(A @ B + bias). Block 256 thr = 8 waves.
// Wave tile 64(M)x32(N) = 4x2 wmma tiles; block tile 128x128.
// EPI: 0 = store f32, 1 = lipswish -> bf16, 2 = relu -> bf16,
//      3 = dual store (f32 D + bf16 D2)
// ---------------------------------------------------------------------------
template <int EPI>
__device__ __forceinline__ void gemm_body(const __bf16* A, int lda,
                                          const __bf16* Bt, int ldb,
                                          const float* bias, void* Dv,
                                          __bf16* D2, int ldd, int K, int row0,
                                          int col0) {
  const int lane = threadIdx.x & 31;
  const int wave = threadIdx.x >> 5;
  const int wm = wave & 1, wn = wave >> 1;
  const int rbase = row0 + wm * 64;
  const int cbase = col0 + wn * 32;

  v8f acc[4][2];
#pragma unroll
  for (int i = 0; i < 4; ++i)
#pragma unroll
    for (int j = 0; j < 2; ++j)
#pragma unroll
      for (int v = 0; v < 8; ++v) acc[i][j][v] = 0.0f;

  for (int k0 = 0; k0 < K; k0 += 32) {
    Frag a[4], b[2];
#pragma unroll
    for (int i = 0; i < 4; ++i)
      load_a(a[i], A + (size_t)(rbase + i * 16) * lda + k0, lda, lane);
#pragma unroll
    for (int j = 0; j < 2; ++j)
      load_b(b[j], Bt + (size_t)(cbase + j * 16) * ldb + k0, ldb, lane);
#pragma unroll
    for (int i = 0; i < 4; ++i)
#pragma unroll
      for (int j = 0; j < 2; ++j) acc[i][j] = wmma_bf16(a[i], b[j], acc[i][j]);
  }

  const int half = lane >> 4, nn = lane & 15;
#pragma unroll
  for (int j = 0; j < 2; ++j) {
    const int col = cbase + j * 16 + nn;
    const float bv = bias ? bias[col] : 0.0f;
#pragma unroll
    for (int i = 0; i < 4; ++i) {
#pragma unroll
      for (int v = 0; v < 8; ++v) {
        const int row = rbase + i * 16 + half * 8 + v;
        const size_t idx = (size_t)row * ldd + col;
        float x = acc[i][j][v] + bv;
        if (EPI == 0) {
          ((float*)Dv)[idx] = x;
        } else if (EPI == 1) {
          float s = 1.0f / (1.0f + __expf(-x));
          ((__bf16*)Dv)[idx] = (__bf16)(0.909f * x * s);
        } else if (EPI == 2) {
          ((__bf16*)Dv)[idx] = (__bf16)(x > 0.0f ? x : 0.0f);
        } else {
          ((float*)Dv)[idx] = x;
          D2[idx] = (__bf16)x;
        }
      }
    }
  }
}

__global__ __launch_bounds__(256) void k_gemm_f32out(
    const __bf16* A, int lda, const __bf16* Bt, int ldb, const float* bias,
    float* D, int ldd, int K) {
  gemm_body<0>(A, lda, Bt, ldb, bias, D, nullptr, ldd, K, blockIdx.y * 128,
               blockIdx.x * 128);
}
__global__ __launch_bounds__(256) void k_gemm_relu(
    const __bf16* A, int lda, const __bf16* Bt, int ldb, const float* bias,
    __bf16* D, int ldd, int K) {
  gemm_body<2>(A, lda, Bt, ldb, bias, D, nullptr, ldd, K, blockIdx.y * 128,
               blockIdx.x * 128);
}
__global__ __launch_bounds__(256) void k_gemm_dual(
    const __bf16* A, int lda, const __bf16* Bt, int ldb, const float* bias,
    float* D, __bf16* D2, int ldd, int K) {
  gemm_body<3>(A, lda, Bt, ldb, bias, D, D2, ldd, K, blockIdx.y * 128,
               blockIdx.x * 128);
}

// Layer-1 of both MLPs: h = lipswish(y16 @ W0 + b0); blockIdx.z selects f/g.
__global__ __launch_bounds__(256) void k_layer1(
    const __bf16* y16, const __bf16* fw0T, const __bf16* gw0T,
    const float* fb0, const float* gb0, __bf16* hf, __bf16* hg) {
  const __bf16* Bt = blockIdx.z ? gw0T : fw0T;
  const float* bias = blockIdx.z ? gb0 : fb0;
  __bf16* D = blockIdx.z ? hg : hf;
  gemm_body<1>(y16, H_DIM, Bt, H_DIM, bias, D, nullptr, H_DIM, H_DIM,
               blockIdx.y * 128, blockIdx.x * 128);
}

// Fused layer-2 drift + diffusion GEMMs + Euler-Maruyama update:
// y += (hf@fw1 + fb1)*dt + (hg@gw1 + gb1)*(dW*sqrt(dt)); also refresh y16.
__global__ __launch_bounds__(256) void k_step2(
    const __bf16* hf, const __bf16* hg, const __bf16* fw1T,
    const __bf16* gw1T, const float* fb1, const float* gb1, const float* dwk,
    float* y, __bf16* y16) {
  const int lane = threadIdx.x & 31;
  const int wave = threadIdx.x >> 5;
  const int wm = wave & 1, wn = wave >> 1;
  const int rbase = blockIdx.y * 128 + wm * 64;
  const int cbase = blockIdx.x * 128 + wn * 32;

  v8f accF[4][2], accG[4][2];
#pragma unroll
  for (int i = 0; i < 4; ++i)
#pragma unroll
    for (int j = 0; j < 2; ++j)
#pragma unroll
      for (int v = 0; v < 8; ++v) {
        accF[i][j][v] = 0.0f;
        accG[i][j][v] = 0.0f;
      }

  // Pass 1: drift
  for (int k0 = 0; k0 < H_DIM; k0 += 32) {
    Frag a[4], b[2];
#pragma unroll
    for (int i = 0; i < 4; ++i)
      load_a(a[i], hf + (size_t)(rbase + i * 16) * H_DIM + k0, H_DIM, lane);
#pragma unroll
    for (int j = 0; j < 2; ++j)
      load_b(b[j], fw1T + (size_t)(cbase + j * 16) * H_DIM + k0, H_DIM, lane);
#pragma unroll
    for (int i = 0; i < 4; ++i)
#pragma unroll
      for (int j = 0; j < 2; ++j) accF[i][j] = wmma_bf16(a[i], b[j], accF[i][j]);
  }
  // Pass 2: diffusion
  for (int k0 = 0; k0 < H_DIM; k0 += 32) {
    Frag a[4], b[2];
#pragma unroll
    for (int i = 0; i < 4; ++i)
      load_a(a[i], hg + (size_t)(rbase + i * 16) * H_DIM + k0, H_DIM, lane);
#pragma unroll
    for (int j = 0; j < 2; ++j)
      load_b(b[j], gw1T + (size_t)(cbase + j * 16) * H_DIM + k0, H_DIM, lane);
#pragma unroll
    for (int i = 0; i < 4; ++i)
#pragma unroll
      for (int j = 0; j < 2; ++j) accG[i][j] = wmma_bf16(a[i], b[j], accG[i][j]);
  }

  const int half = lane >> 4, nn = lane & 15;
#pragma unroll
  for (int j = 0; j < 2; ++j) {
    const int col = cbase + j * 16 + nn;
    const float bf = fb1[col];
    const float bg = gb1[col];
#pragma unroll
    for (int i = 0; i < 4; ++i) {
#pragma unroll
      for (int v = 0; v < 8; ++v) {
        const int row = rbase + i * 16 + half * 8 + v;
        const size_t idx = (size_t)row * H_DIM + col;
        float drift = accF[i][j][v] + bf;
        float diff = accG[i][j][v] + bg;
        float yn = y[idx] + drift * DT_C + diff * (dwk[idx] * SQRT_DT_C);
        y[idx] = yn;
        y16[idx] = (__bf16)yn;
      }
    }
  }
}

// One-time helpers -----------------------------------------------------------
__global__ void k_transpose_bf16(const float* __restrict__ src,   // [K,N]
                                 __bf16* __restrict__ dst,        // [N,K]
                                 int K, int N) {
  int idx = blockIdx.x * blockDim.x + threadIdx.x;
  if (idx >= K * N) return;
  int k = idx / N, n = idx - k * N;
  dst[(size_t)n * K + k] = (__bf16)src[idx];
}

// coeffs[:,0,:] -> bf16 [B, D] (row stride T*D in the source)
__global__ void k_coeffs_bf16(const float* __restrict__ coeffs,
                              __bf16* __restrict__ c16) {
  int idx = blockIdx.x * blockDim.x + threadIdx.x;
  if (idx >= B_DIM * D_DIM) return;
  int b = idx / D_DIM, d = idx - b * D_DIM;
  c16[idx] = (__bf16)coeffs[(size_t)b * (T_DIM * D_DIM) + d];
}

__global__ void k_tfill(__bf16* cat) {  // cat[b][128+j] = j/127
  int idx = blockIdx.x * blockDim.x + threadIdx.x;
  if (idx >= B_DIM * FH_DIM) return;
  int b = idx >> 7, j = idx & 127;
  cat[(size_t)b * 256 + 128 + j] = (__bf16)((float)j * (1.0f / 127.0f));
}

// ---------------------------------------------------------------------------
extern "C" void kernel_launch(void* const* d_in, const int* in_sizes, int n_in,
                              void* d_out, int out_size, void* d_ws,
                              size_t ws_size, hipStream_t stream) {
  const float* coeffs    = (const float*)d_in[0];
  // d_in[1] = times (unused; knots are the coeffs themselves)
  const float* dw        = (const float*)d_in[2];
  const float* f_ws      = (const float*)d_in[3];
  const float* f_bs      = (const float*)d_in[4];
  const float* g_ws      = (const float*)d_in[5];
  const float* g_bs      = (const float*)d_in[6];
  const float* initial_w = (const float*)d_in[7];
  const float* initial_b = (const float*)d_in[8];
  const float* lin_in_w  = (const float*)d_in[9];
  const float* lin_in_b  = (const float*)d_in[10];
  const float* emb_w     = (const float*)d_in[11];
  const float* emb_b     = (const float*)d_in[12];
  const float* samp_w    = (const float*)d_in[13];
  const float* samp_b    = (const float*)d_in[14];

  __bf16* ws16 = (__bf16*)d_ws;
  size_t off = 0;
  auto take = [&](size_t n) { __bf16* p = ws16 + off; off += n; return p; };
  __bf16* fw0T  = take((size_t)H_DIM * H_DIM);
  __bf16* fw1T  = take((size_t)H_DIM * H_DIM);
  __bf16* gw0T  = take((size_t)H_DIM * H_DIM);
  __bf16* gw1T  = take((size_t)H_DIM * H_DIM);
  __bf16* initT = take((size_t)H_DIM * D_DIM);
  __bf16* linT  = take((size_t)FH_DIM * H_DIM);
  __bf16* embT  = take((size_t)H_DIM * 256);
  __bf16* sampT = take((size_t)FH_DIM * H_DIM);
  __bf16* hf    = take((size_t)B_DIM * H_DIM);
  __bf16* hg    = take((size_t)B_DIM * H_DIM);
  __bf16* cat   = take((size_t)B_DIM * 256);
  __bf16* z2    = take((size_t)B_DIM * H_DIM);
  __bf16* y16   = take((size_t)B_DIM * H_DIM);
  __bf16* c16   = take((size_t)B_DIM * D_DIM);
  float* y = (float*)(ws16 + off);

  const dim3 blk(256);
  auto tr = [&](const float* s, __bf16* d, int K, int N) {
    int total = K * N;
    k_transpose_bf16<<<(total + 255) / 256, 256, 0, stream>>>(s, d, K, N);
  };
  tr(f_ws,                 fw0T, H_DIM, H_DIM);
  tr(f_ws + H_DIM * H_DIM, fw1T, H_DIM, H_DIM);
  tr(g_ws,                 gw0T, H_DIM, H_DIM);
  tr(g_ws + H_DIM * H_DIM, gw1T, H_DIM, H_DIM);
  tr(initial_w, initT, D_DIM, H_DIM);
  tr(lin_in_w,  linT,  H_DIM, FH_DIM);
  tr(emb_w,     embT,  256,   H_DIM);
  tr(samp_w,    sampT, H_DIM, FH_DIM);
  k_coeffs_bf16<<<(B_DIM * D_DIM + 255) / 256, 256, 0, stream>>>(coeffs, c16);
  k_tfill<<<(B_DIM * FH_DIM + 255) / 256, 256, 0, stream>>>(cat);

  // y0 = coeffs[:,0,:] @ initial_w + initial_b  (writes fp32 y and bf16 y16)
  k_gemm_dual<<<dim3(8, 8), blk, 0, stream>>>(c16, D_DIM, initT, D_DIM,
                                              initial_b, y, y16, H_DIM, D_DIM);

  for (int s = 0; s < NSTEPS; ++s) {
    k_layer1<<<dim3(8, 8, 2), blk, 0, stream>>>(y16, fw0T, gw0T, f_bs, g_bs,
                                                hf, hg);
    k_step2<<<dim3(8, 8), blk, 0, stream>>>(hf, hg, fw1T, gw1T, f_bs + H_DIM,
                                            g_bs + H_DIM,
                                            dw + (size_t)s * B_DIM * H_DIM, y,
                                            y16);
  }

  // z = relu(y @ lin_in_w + b) -> cat[:, :128]
  k_gemm_relu<<<dim3(1, 8), blk, 0, stream>>>(y16, H_DIM, linT, H_DIM,
                                              lin_in_b, cat, 256, H_DIM);
  // z = relu(cat @ emb_w + b)
  k_gemm_relu<<<dim3(8, 8), blk, 0, stream>>>(cat, 256, embT, 256, emb_b, z2,
                                              H_DIM, 256);
  // out = z @ samp_w + b   (fp32)
  k_gemm_f32out<<<dim3(1, 8), blk, 0, stream>>>(z2, H_DIM, sampT, H_DIM,
                                                samp_b, (float*)d_out, FH_DIM,
                                                H_DIM);
}
